// KGAN_32873679683604
// MI455X (gfx1250) — matrix operand order repaired and checked by reading, not compiled, not run on hardware
//
#include <hip/hip_runtime.h>
#include <hip/hip_bf16.h>

// ---- problem constants (from reference) ----
#define BATCH   1024
#define GG      16        // G = N_REL - 1
#define NMEM    32
#define DIM     64
#define NHOPS   2
#define NROWS   100001u   // entity table rows (N_ENT + 1)

typedef __attribute__((ext_vector_type(16))) __bf16       v16bf;
typedef __attribute__((ext_vector_type(8)))  float        v8f;
typedef __attribute__((ext_vector_type(4)))  unsigned int v4u;
typedef __attribute__((ext_vector_type(8)))  unsigned int v8u;

// ---------------------------------------------------------------------------
// TDM gather-mode descriptor + issue (CDNA5 ISA/§8).
// One descriptor gathers 8 rows (32-bit indices) of `tile_dim0`=64 f32
// elements from a row-major table straight into LDS (2 KB contiguous).
// Issued wave-uniformly; tracked with TENSORcnt.
// ---------------------------------------------------------------------------
__device__ __forceinline__ unsigned int lds_off_u32(const void* p) {
    // low 32 bits of the flat LDS-aperture address == byte offset in wave LDS
    return (unsigned int)(uintptr_t)p;
}

__device__ __forceinline__ void tdm_gather8_rows(const float* table,
                                                 unsigned int lds_byte,
                                                 const int* idx /*[8]*/) {
    const unsigned long long ga = (unsigned long long)(uintptr_t)table;

    v4u g0;
    g0[0] = 0xC0000001u;                               // count=1 | idx32 | gather_mode
    g0[1] = lds_byte;                                  // lds_addr
    g0[2] = (unsigned int)(ga & 0xFFFFFFFFu);          // global_addr[31:0]
    g0[3] = (unsigned int)((ga >> 32) & 0x1FFFFFFu)    // global_addr[56:32]
          | (2u << 30);                                // type = 2 ("image")

    v8u g1;
    g1[0] = (2u << 16);                                // data_size = 4 B; mask=0
    g1[1] = (DIM & 0xFFFFu) << 16;                     // tensor_dim0[15:0] = 64
    g1[2] = (NROWS & 0xFFFFu) << 16;                   // dim0 hi=0 | tensor_dim1[15:0]
    g1[3] = ((NROWS >> 16) & 0xFFFFu)                  // tensor_dim1[31:16]
          | ((unsigned)DIM << 16);                     // tile_dim0 = 64
    g1[4] = 8u;                                        // tile_dim1 = 8 valid indices
    g1[5] = (unsigned)DIM;                             // tensor_dim0_stride = 64
    g1[6] = 0u;
    g1[7] = 0u;

    v4u g2, g3;
    g2[0] = (unsigned)idx[0]; g2[1] = (unsigned)idx[1];
    g2[2] = (unsigned)idx[2]; g2[3] = (unsigned)idx[3];
    g3[0] = (unsigned)idx[4]; g3[1] = (unsigned)idx[5];
    g3[2] = (unsigned)idx[6]; g3[3] = (unsigned)idx[7];

    asm volatile("tensor_load_to_lds %0, %1, %2, %3"
                 :
                 : "s"(g0), "s"(g1), "s"(g2), "s"(g3)
                 : "memory");
}

// ---------------------------------------------------------------------------
// Kernel 0: zero accumulators + y buffer
// ---------------------------------------------------------------------------
__global__ void kgan_zero_kernel(float* acc, float* y_buf) {
    int i = blockIdx.x * 256 + threadIdx.x;
    if (i < 8) acc[i] = 0.0f;
    if (i < BATCH * DIM) y_buf[i] = 0.0f;
}

// ---------------------------------------------------------------------------
// Kernel 1: per-(b,g) fused memory hop.
// h/t tiles [32x64] DMA'd into LDS via TDM gather mode (wave 0 issues 8
// descriptors, waits TENSORcnt==0); r/v gathered on the normal load path.
// Then rv, reshape(-1,32) softmax, o-reduction, kge sigmoid sum, l2 squares.
// ---------------------------------------------------------------------------
__global__ __launch_bounds__(256)
void kgan_hop_kernel(const int* __restrict__ mem_h, const int* __restrict__ mem_r,
                     const int* __restrict__ mem_t, const int* __restrict__ pos_items,
                     const float* __restrict__ entity_emb,
                     const float* __restrict__ relation_emb,
                     float* __restrict__ o_buf, float* __restrict__ acc) {
    const int bg  = blockIdx.x;          // 0 .. B*G-1
    const int b   = bg >> 4;
    const int tid = threadIdx.x;

    __shared__ float sh[NMEM * DIM];
    __shared__ float sr[NMEM * DIM];
    __shared__ float st[NMEM * DIM];
    __shared__ float sprob[NMEM * DIM];
    __shared__ float sv[DIM];
    __shared__ float srv[NMEM];
    __shared__ float red[256];

    const int base = bg * NMEM;

    // --- wave 0: issue TDM gathers for h and t (4 descriptors each) ---
    if (tid < 32) {
        for (int dsc = 0; dsc < 4; ++dsc) {
            int idxh[8], idxt[8];
#pragma unroll
            for (int j = 0; j < 8; ++j) {
                idxh[j] = mem_h[base + dsc * 8 + j];   // block-uniform scalar loads
                idxt[j] = mem_t[base + dsc * 8 + j];
            }
            tdm_gather8_rows(entity_emb, lds_off_u32(&sh[dsc * 8 * DIM]), idxh);
            tdm_gather8_rows(entity_emb, lds_off_u32(&st[dsc * 8 * DIM]), idxt);
        }
    }

    // --- meanwhile: r gather (tiny table) + v ---
    for (int i = tid; i < NMEM * DIM; i += 256) {
        int m = i >> 6, d = i & 63;
        sr[i] = relation_emb[(size_t)mem_r[base + m] * DIM + d];
    }
    if (tid < DIM) sv[tid] = entity_emb[(size_t)pos_items[b] * DIM + tid];

    if (tid < 32) __builtin_amdgcn_s_wait_tensorcnt(0);   // DMA complete
    __syncthreads();

    // rv[m] = dot(r[m], v)
    if (tid < NMEM) {
        float s = 0.0f;
        for (int d = 0; d < DIM; ++d) s += sr[tid * DIM + d] * sv[d];
        srv[tid] = s;
    }
    __syncthreads();

    // softmax over each contiguous 32-chunk of d (faithful reshape(-1, 32))
    if (tid < 2 * NMEM) {
        int m = tid >> 1, half = tid & 1;
        const float rv = srv[m];
        const int off = m * DIM + half * 32;
        float mx = -3.4e38f;
        for (int j = 0; j < 32; ++j) { float x = sh[off + j] * rv; mx = fmaxf(mx, x); }
        float ssum = 0.0f;
        for (int j = 0; j < 32; ++j) {
            float e = __expf(sh[off + j] * rv - mx);
            sprob[off + j] = e; ssum += e;
        }
        float inv = 1.0f / ssum;
        for (int j = 0; j < 32; ++j) sprob[off + j] *= inv;
    }
    __syncthreads();

    // o[d] = sum_m t[m,d] * probs[m,d]
    if (tid < DIM) {
        float s = 0.0f;
        for (int m = 0; m < NMEM; ++m) s += st[m * DIM + tid] * sprob[m * DIM + tid];
        o_buf[(size_t)bg * DIM + tid] = s;
    }
    __syncthreads();

    // rt[m] = dot(r[m], t[m])   (reuse srv)
    if (tid < NMEM) {
        float s = 0.0f;
        for (int d = 0; d < DIM; ++d) s += sr[tid * DIM + d] * st[tid * DIM + d];
        srv[tid] = s;
    }
    __syncthreads();

    float kge_p = 0.0f, l2_p = 0.0f;
    for (int i = tid; i < NMEM * DIM; i += 256) {
        int m = i >> 6;
        float x = sh[i] * srv[m];
        kge_p += 1.0f / (1.0f + __expf(-x));
        l2_p  += sh[i] * sh[i] + st[i] * st[i] + sr[i] * sr[i];
    }

    red[tid] = kge_p; __syncthreads();
    for (int s = 128; s > 0; s >>= 1) { if (tid < s) red[tid] += red[tid + s]; __syncthreads(); }
    if (tid == 0) atomicAdd(&acc[0], red[0]);
    __syncthreads();
    red[tid] = l2_p; __syncthreads();
    for (int s = 128; s > 0; s >>= 1) { if (tid < s) red[tid] += red[tid + s]; __syncthreads(); }
    if (tid == 0) atomicAdd(&acc[1], red[0]);
}

// ---------------------------------------------------------------------------
// WMMA fragment lane layouts (16-bit A 16x32 MxK / B 32x16 KxN, wave32):
//   sub = lane>>4, m/n = lane&15
//   A vgpr-pair j: K = kc*32 + (j>>2)*16 + sub*8 + (j&3)*2 , M = lane&15
//   B vgpr-pair j: K = kc*32 + (j>>2)*16 + sub*8 + (j&3)*2 , N = lane&15
//   C/D: lane holds rows (j + sub*8), col = lane&15
// ---------------------------------------------------------------------------

// ---------------------------------------------------------------------------
// Kernel 2: attention.  W1 staged in LDS once per block.  One wave per b:
// hid = o[b] @ W1 via v_wmma_f32_16x16x32_bf16, a = relu(hid @ w2),
// softmax over G, o_h = sum_g w[g]*o[g,:].  y += o_h.
// ---------------------------------------------------------------------------
__global__ __launch_bounds__(256)
void kgan_attn_kernel(const float* __restrict__ o_buf,
                      const float* __restrict__ w1, const float* __restrict__ w2,
                      float* __restrict__ oh_buf, float* __restrict__ y_buf) {
    const int tid  = threadIdx.x;
    const int wave = tid >> 5;
    const int lane = tid & 31;
    const int b    = blockIdx.x * 8 + wave;   // grid = B/8

    __shared__ float sw1[DIM * DIM];          // 16 KB, shared by 8 waves
    __shared__ float sw2[DIM];
    __shared__ float shid[8][GG * DIM];       // 32 KB
    __shared__ float sa[8][GG];

    for (int i = tid; i < DIM * DIM; i += 256) sw1[i] = w1[i];
    if (tid < DIM) sw2[tid] = w2[tid];
    __syncthreads();

    const float* o = o_buf + (size_t)b * (GG * DIM);
    const int sub = lane >> 4;
    const int n   = lane & 15;

#pragma unroll
    for (int nt = 0; nt < 4; ++nt) {
        v8f c = {};
#pragma unroll
        for (int kc = 0; kc < 2; ++kc) {
            v16bf a, bb;
#pragma unroll
            for (int j = 0; j < 8; ++j) {
                int k = kc * 32 + (j >> 2) * 16 + sub * 8 + (j & 3) * 2;
                a[2 * j]     = (__bf16)o[n * DIM + k];
                a[2 * j + 1] = (__bf16)o[n * DIM + k + 1];
            }
#pragma unroll
            for (int j = 0; j < 8; ++j) {
                int k = kc * 32 + (j >> 2) * 16 + sub * 8 + (j & 3) * 2;
                bb[2 * j]     = (__bf16)sw1[k * DIM + nt * 16 + n];
                bb[2 * j + 1] = (__bf16)sw1[(k + 1) * DIM + nt * 16 + n];
            }
            c = __builtin_amdgcn_wmma_f32_16x16x32_bf16(false, a, false, bb,
                                                        (short)0, c, false, false);
        }
#pragma unroll
        for (int j = 0; j < 8; ++j)
            shid[wave][(j + sub * 8) * DIM + nt * 16 + n] = c[j];
    }
    __syncthreads();

    // a[g] = relu(dot(hid[g,:], w2))
    if (lane < GG) {
        float s = 0.0f;
        for (int e = 0; e < DIM; ++e) s += shid[wave][lane * DIM + e] * sw2[e];
        sa[wave][lane] = s > 0.0f ? s : 0.0f;
    }
    __syncthreads();

    // softmax over G (redundantly in every lane)
    float mx = -3.4e38f;
    for (int g = 0; g < GG; ++g) mx = fmaxf(mx, sa[wave][g]);
    float wgt[GG];
    float ssum = 0.0f;
    for (int g = 0; g < GG; ++g) { wgt[g] = __expf(sa[wave][g] - mx); ssum += wgt[g]; }
    float inv = 1.0f / ssum;

    for (int dd = lane; dd < DIM; dd += 32) {
        float s = 0.0f;
        for (int g = 0; g < GG; ++g) s += o[g * DIM + dd] * wgt[g];
        s *= inv;
        oh_buf[(size_t)b * DIM + dd] = s;
        y_buf[(size_t)b * DIM + dd] += s;   // each b owned by one wave: race-free
    }
}

// ---------------------------------------------------------------------------
// Kernel 3: items_emb_new = (v + o_h_last) @ T via WMMA (one wave per 16
// batch rows, T staged in LDS), then scores/neg_scores dot with y and the
// log-sigmoid partial sum.
// ---------------------------------------------------------------------------
__global__ __launch_bounds__(256)
void kgan_final_kernel(const float* __restrict__ entity_emb,
                       const int* __restrict__ pos_items, const int* __restrict__ neg_items,
                       const float* __restrict__ oh_last, const float* __restrict__ transform,
                       const float* __restrict__ y_buf, float* __restrict__ acc) {
    const int tid  = threadIdx.x;
    const int wave = tid >> 5;
    const int lane = tid & 31;
    const int tile = blockIdx.x * 8 + wave;   // 64 tiles x 16 rows
    const int row0 = tile * 16;

    __shared__ float sT[DIM * DIM];           // 16 KB
    __shared__ float sx[8][16 * DIM];         // v + o_h
    __shared__ float snew[8][16 * DIM];
    __shared__ float red[256];

    for (int i = tid; i < DIM * DIM; i += 256) sT[i] = transform[i];

    const int sub = lane >> 4;
    const int n   = lane & 15;

    for (int i = lane; i < 16 * DIM; i += 32) {
        int rm = i >> 6, d = i & 63;
        int bidx = row0 + rm;
        sx[wave][i] = entity_emb[(size_t)pos_items[bidx] * DIM + d]
                    + oh_last[(size_t)bidx * DIM + d];
    }
    __syncthreads();

#pragma unroll
    for (int nt = 0; nt < 4; ++nt) {
        v8f c = {};
#pragma unroll
        for (int kc = 0; kc < 2; ++kc) {
            v16bf a, bb;
#pragma unroll
            for (int j = 0; j < 8; ++j) {
                int k = kc * 32 + (j >> 2) * 16 + sub * 8 + (j & 3) * 2;
                a[2 * j]     = (__bf16)sx[wave][n * DIM + k];
                a[2 * j + 1] = (__bf16)sx[wave][n * DIM + k + 1];
            }
#pragma unroll
            for (int j = 0; j < 8; ++j) {
                int k = kc * 32 + (j >> 2) * 16 + sub * 8 + (j & 3) * 2;
                bb[2 * j]     = (__bf16)sT[k * DIM + nt * 16 + n];
                bb[2 * j + 1] = (__bf16)sT[(k + 1) * DIM + nt * 16 + n];
            }
            c = __builtin_amdgcn_wmma_f32_16x16x32_bf16(false, a, false, bb,
                                                        (short)0, c, false, false);
        }
#pragma unroll
        for (int j = 0; j < 8; ++j)
            snew[wave][(j + sub * 8) * DIM + nt * 16 + n] = c[j];
    }
    __syncthreads();

    float part = 0.0f;
    if (lane < 16) {
        int bidx = row0 + lane;
        const float* yv = y_buf + (size_t)bidx * DIM;
        const float* ne = entity_emb + (size_t)neg_items[bidx] * DIM;
        float sc = 0.0f, nsc = 0.0f;
        for (int d = 0; d < DIM; ++d) {
            sc  += snew[wave][lane * DIM + d] * yv[d];
            nsc += ne[d] * yv[d];
        }
        float x = sc - nsc;
        part = (x > 0.0f) ? -__logf(1.0f + __expf(-x))
                          : (x - __logf(1.0f + __expf(x)));
    }
    red[tid] = part; __syncthreads();
    for (int s = 128; s > 0; s >>= 1) { if (tid < s) red[tid] += red[tid + s]; __syncthreads(); }
    if (tid == 0) atomicAdd(&acc[2], red[0]);
}

// ---------------------------------------------------------------------------
// Kernel 4: combine scalars.
// ---------------------------------------------------------------------------
__global__ void kgan_finalize_kernel(const float* __restrict__ acc, float* __restrict__ out) {
    const float denom = (float)((long long)BATCH * GG * NMEM * DIM);  // per-hop mean denom
    float kge = acc[0] / denom;          // sum over hops of per-hop means
    float l2  = acc[1];
    float mf  = -acc[2] / (float)BATCH;
    out[0] = mf - 0.01f * kge + 1e-5f * l2;
}

// ---------------------------------------------------------------------------
extern "C" void kernel_launch(void* const* d_in, const int* in_sizes, int n_in,
                              void* d_out, int out_size, void* d_ws, size_t ws_size,
                              hipStream_t stream) {
    const int*   pos_items    = (const int*)d_in[0];
    const int*   neg_items    = (const int*)d_in[1];
    const int*   mem_h        = (const int*)d_in[2];
    const int*   mem_r        = (const int*)d_in[3];
    const int*   mem_t        = (const int*)d_in[4];
    const float* entity_emb   = (const float*)d_in[5];
    const float* relation_emb = (const float*)d_in[6];
    const float* transform    = (const float*)d_in[7];
    const float* attn_w1      = (const float*)d_in[8];
    const float* attn_w2      = (const float*)d_in[9];

    char*  ws     = (char*)d_ws;
    float* acc    = (float*)ws;                          // 8 floats
    float* o_buf  = (float*)(ws + 256);                  // B*G*D floats (4 MB)
    float* oh_buf = o_buf  + (size_t)BATCH * GG * DIM;   // B*D floats
    float* y_buf  = oh_buf + (size_t)BATCH * DIM;        // B*D floats

    kgan_zero_kernel<<<(BATCH * DIM + 255) / 256, 256, 0, stream>>>(acc, y_buf);

    const size_t hop_stride = (size_t)BATCH * GG * NMEM;
    for (int hop = 0; hop < NHOPS; ++hop) {
        kgan_hop_kernel<<<BATCH * GG, 256, 0, stream>>>(
            mem_h + hop * hop_stride, mem_r + hop * hop_stride, mem_t + hop * hop_stride,
            pos_items, entity_emb, relation_emb, o_buf, acc);
        kgan_attn_kernel<<<BATCH / 8, 256, 0, stream>>>(
            o_buf, attn_w1 + (size_t)hop * DIM * DIM, attn_w2 + (size_t)hop * DIM,
            oh_buf, y_buf);
    }

    kgan_final_kernel<<<BATCH / 16 / 8, 256, 0, stream>>>(
        entity_emb, pos_items, neg_items, oh_buf, transform, y_buf, acc);

    kgan_finalize_kernel<<<1, 1, 0, stream>>>(acc, (float*)d_out);
}
